// RGCNModel_71107478553206
// MI455X (gfx1250) — matrix operand order repaired and checked by reading, not compiled, not run on hardware
//
#include <hip/hip_runtime.h>

// RGCN (2 layers) for MI455X / gfx1250, wave32, bf16 WMMA.
// zero -> fp32 edge scatter (atomics) -> mean+cvt->bf16 agg -> cvt x_dst ->
// pack weights (bf16 B-fragment panels) -> WMMA bf16 GEMM1 (+bias,ReLU, bf16 h)
// -> WMMA bf16 GEMM2 (+bias, fp32 out).
// agg (M x 2048) is shared by both layers (identical gather/scatter in the ref).

#define D_CH 256
#define NREL 8
#define K0LEN (NREL * D_CH) /* 2048 */
#define KTOT (K0LEN + D_CH) /* 2304 */

typedef __bf16 bf16;
typedef bf16 v4bf  __attribute__((ext_vector_type(4)));
typedef bf16 v8bf  __attribute__((ext_vector_type(8)));
typedef bf16 v16bf __attribute__((ext_vector_type(16)));
typedef float v8f  __attribute__((ext_vector_type(8)));

// ---------------- zero workspace (sums + cnt) ----------------
__global__ __launch_bounds__(256) void k_zero(float* __restrict__ p, unsigned n) {
  unsigned idx = (blockIdx.x * 256u + threadIdx.x) * 4u;
  if (idx < n) *(float4*)(p + idx) = make_float4(0.f, 0.f, 0.f, 0.f);
}

// ---------------- edge scatter: sums[seg, :] += x_src[src, :]; cnt[seg]++ ----
__global__ __launch_bounds__(256) void k_scatter(const float* __restrict__ xsrc,
                                                 const int* __restrict__ eidx,
                                                 const int* __restrict__ etyp,
                                                 float* __restrict__ sums,
                                                 float* __restrict__ cnt, int E) {
  const int t = threadIdx.x; // channel 0..255
  const int e0 = blockIdx.x * 4;
#pragma unroll
  for (int i = 0; i < 4; ++i) {
    int e = e0 + i;
    if (e >= E) return;
    int src = eidx[e];
    int dst = eidx[E + e];
    int r = etyp[e];
    int seg = dst * NREL + r;
    float v = xsrc[(size_t)src * D_CH + t];
    atomicAdd(sums + (size_t)seg * D_CH + t, v);
    if (t == 0) atomicAdd(cnt + seg, 1.0f);
  }
}

// ---------------- mean + convert: aggb = bf16(sums / max(cnt,1)) ------------
__global__ __launch_bounds__(256) void k_div_cvt(const float* __restrict__ sums,
                                                 const float* __restrict__ cnt,
                                                 bf16* __restrict__ aggb,
                                                 unsigned nseg) {
  unsigned idx = (blockIdx.x * 256u + threadIdx.x) * 4u;
  if (idx >= nseg * (unsigned)D_CH) return;
  float c = cnt[idx >> 8]; // 256 channels per segment
  float inv = 1.0f / fmaxf(c, 1.0f);
  float4 v = *(const float4*)(sums + idx);
  v4bf o;
  o[0] = (bf16)(v.x * inv); o[1] = (bf16)(v.y * inv);
  o[2] = (bf16)(v.z * inv); o[3] = (bf16)(v.w * inv);
  *(v4bf*)(aggb + idx) = o;
}

// ---------------- fp32 -> bf16 elementwise (x_dst) ----------------
__global__ __launch_bounds__(256) void k_cvt(const float* __restrict__ x,
                                             bf16* __restrict__ y, unsigned n) {
  unsigned idx = (blockIdx.x * 256u + threadIdx.x) * 4u;
  if (idx >= n) return;
  float4 v = *(const float4*)(x + idx);
  v4bf o;
  o[0] = (bf16)v.x; o[1] = (bf16)v.y; o[2] = (bf16)v.z; o[3] = (bf16)v.w;
  *(v4bf*)(y + idx) = o;
}

// ---------------- pack B = [W ; root] into bf16 B-fragment panels -----------
// Panel = 32 K-rows. Element offset: (k>>5)*32*256 + col*32 + (k&31).
// A lane's B fragment (K = k0+16*half .. +15 at column col) is then the 16
// contiguous bf16 at panel + col*32 + 16*half -> one 32-byte load.
__global__ __launch_bounds__(256) void k_packB(const float* __restrict__ W,
                                               const float* __restrict__ root,
                                               bf16* __restrict__ Bp) {
  unsigned idx = blockIdx.x * 256u + threadIdx.x; // < KTOT*256
  if (idx >= (unsigned)KTOT * D_CH) return;
  unsigned k = idx >> 8;     // row
  unsigned col = idx & 255u; // col
  float v = (k < (unsigned)K0LEN) ? W[(size_t)k * D_CH + col]
                                  : root[(size_t)(k - K0LEN) * D_CH + col];
  Bp[(size_t)(k >> 5) * (32 * D_CH) + col * 32 + (k & 31u)] = (bf16)v;
}

// A fragment (16x32 bf16; contiguous k order used here, not the ISA table order): lane row m = lane%16;
// elements 0..15 = K = k0 + 16*half + 0..15 (matches packed B).
__device__ inline v16bf load_afrag(const bf16* __restrict__ ar, int k0, int half) {
  const bf16* p = ar + k0 + 16 * half;
  v8bf lo = *(const v8bf*)(p);
  v8bf hi = *(const v8bf*)(p + 8);
  return __builtin_shufflevector(lo, hi, 0, 1, 2, 3, 4, 5, 6, 7,
                                 8, 9, 10, 11, 12, 13, 14, 15);
}

// ---------------- fused GEMM: C = [A0|A1] @ Bp + bias (opt ReLU) ------------
// A0: M x 2048 bf16 (agg), A1: M x 256 bf16 (x_dst or h); Bp: packed KTOT x 256.
// Block = 128 threads (4 waves); 32-row M slab (two 16-row tiles); wave owns a
// 64-col N strip -> 8 v8f accumulators. Per K-step(32): 4 A b128 + 8 B b128
// loads + 8 v_wmma_f32_16x16x32_bf16 (16384 FLOP each).
template <int RELU, typename OutT>
__global__ __launch_bounds__(128) void k_gemm(const bf16* __restrict__ A0,
                                              const bf16* __restrict__ A1,
                                              const bf16* __restrict__ Bp,
                                              const float* __restrict__ bias,
                                              OutT* __restrict__ C, int M) {
  const int lane = threadIdx.x & 31;
  const int wave = threadIdx.x >> 5; // 0..3 -> N strip of 64
  const int half = lane >> 4;        // lane-half
  const int lrow = lane & 15;
  const int nbase = wave * 64;
  const int c0 = nbase + lrow;

  int r0 = blockIdx.x * 32 + lrow; // tile 0 A row
  int r1 = r0 + 16;                // tile 1 A row
  if (r0 >= M) r0 = M - 1;         // branchless clamp keeps EXEC all-ones
  if (r1 >= M) r1 = M - 1;

  v8f a00 = {}, a01 = {}, a02 = {}, a03 = {};
  v8f a10 = {}, a11 = {}, a12 = {}, a13 = {};

  // ---- phase 0: agg @ W (panels 0..63) ----
  {
    const bf16* ar0 = A0 + (size_t)r0 * K0LEN;
    const bf16* ar1 = A0 + (size_t)r1 * K0LEN;
    for (int k0 = 0; k0 < K0LEN; k0 += 32) {
      v16bf fa0 = load_afrag(ar0, k0, half);
      v16bf fa1 = load_afrag(ar1, k0, half);
      const bf16* bp = Bp + (size_t)(k0 >> 5) * (32 * D_CH) + (size_t)c0 * 32 + 16 * half;
      v16bf b0 = *(const v16bf*)(bp + 0);
      v16bf b1 = *(const v16bf*)(bp + 512);
      v16bf b2 = *(const v16bf*)(bp + 1024);
      v16bf b3 = *(const v16bf*)(bp + 1536);
      a00 = __builtin_amdgcn_wmma_f32_16x16x32_bf16(false, fa0, false, b0, (short)0, a00, false, false);
      a01 = __builtin_amdgcn_wmma_f32_16x16x32_bf16(false, fa0, false, b1, (short)0, a01, false, false);
      a02 = __builtin_amdgcn_wmma_f32_16x16x32_bf16(false, fa0, false, b2, (short)0, a02, false, false);
      a03 = __builtin_amdgcn_wmma_f32_16x16x32_bf16(false, fa0, false, b3, (short)0, a03, false, false);
      a10 = __builtin_amdgcn_wmma_f32_16x16x32_bf16(false, fa1, false, b0, (short)0, a10, false, false);
      a11 = __builtin_amdgcn_wmma_f32_16x16x32_bf16(false, fa1, false, b1, (short)0, a11, false, false);
      a12 = __builtin_amdgcn_wmma_f32_16x16x32_bf16(false, fa1, false, b2, (short)0, a12, false, false);
      a13 = __builtin_amdgcn_wmma_f32_16x16x32_bf16(false, fa1, false, b3, (short)0, a13, false, false);
    }
  }
  // ---- phase 1: x @ root (panels 64..71) ----
  {
    const bf16* ar0 = A1 + (size_t)r0 * D_CH;
    const bf16* ar1 = A1 + (size_t)r1 * D_CH;
    const bf16* BpR = Bp + (size_t)(K0LEN >> 5) * (32 * D_CH);
    for (int k0 = 0; k0 < D_CH; k0 += 32) {
      v16bf fa0 = load_afrag(ar0, k0, half);
      v16bf fa1 = load_afrag(ar1, k0, half);
      const bf16* bp = BpR + (size_t)(k0 >> 5) * (32 * D_CH) + (size_t)c0 * 32 + 16 * half;
      v16bf b0 = *(const v16bf*)(bp + 0);
      v16bf b1 = *(const v16bf*)(bp + 512);
      v16bf b2 = *(const v16bf*)(bp + 1024);
      v16bf b3 = *(const v16bf*)(bp + 1536);
      a00 = __builtin_amdgcn_wmma_f32_16x16x32_bf16(false, fa0, false, b0, (short)0, a00, false, false);
      a01 = __builtin_amdgcn_wmma_f32_16x16x32_bf16(false, fa0, false, b1, (short)0, a01, false, false);
      a02 = __builtin_amdgcn_wmma_f32_16x16x32_bf16(false, fa0, false, b2, (short)0, a02, false, false);
      a03 = __builtin_amdgcn_wmma_f32_16x16x32_bf16(false, fa0, false, b3, (short)0, a03, false, false);
      a10 = __builtin_amdgcn_wmma_f32_16x16x32_bf16(false, fa1, false, b0, (short)0, a10, false, false);
      a11 = __builtin_amdgcn_wmma_f32_16x16x32_bf16(false, fa1, false, b1, (short)0, a11, false, false);
      a12 = __builtin_amdgcn_wmma_f32_16x16x32_bf16(false, fa1, false, b2, (short)0, a12, false, false);
      a13 = __builtin_amdgcn_wmma_f32_16x16x32_bf16(false, fa1, false, b3, (short)0, a13, false, false);
    }
  }

  // ---- epilogue: D layout VGPR i -> row i (lanes 0-15) / i+8 (lanes 16-31) --
  const float bv0 = bias[nbase + 0 + lrow];
  const float bv1 = bias[nbase + 16 + lrow];
  const float bv2 = bias[nbase + 32 + lrow];
  const float bv3 = bias[nbase + 48 + lrow];
#pragma unroll
  for (int i = 0; i < 8; ++i) {
    int row0 = blockIdx.x * 32 + i + 8 * half;
    int row1 = row0 + 16;
    float v;
    if (row0 < M) {
      OutT* crow = C + (size_t)row0 * D_CH;
      v = a00[i] + bv0; if (RELU) v = fmaxf(v, 0.f); crow[nbase + 0 + lrow] = (OutT)v;
      v = a01[i] + bv1; if (RELU) v = fmaxf(v, 0.f); crow[nbase + 16 + lrow] = (OutT)v;
      v = a02[i] + bv2; if (RELU) v = fmaxf(v, 0.f); crow[nbase + 32 + lrow] = (OutT)v;
      v = a03[i] + bv3; if (RELU) v = fmaxf(v, 0.f); crow[nbase + 48 + lrow] = (OutT)v;
    }
    if (row1 < M) {
      OutT* crow = C + (size_t)row1 * D_CH;
      v = a10[i] + bv0; if (RELU) v = fmaxf(v, 0.f); crow[nbase + 0 + lrow] = (OutT)v;
      v = a11[i] + bv1; if (RELU) v = fmaxf(v, 0.f); crow[nbase + 16 + lrow] = (OutT)v;
      v = a12[i] + bv2; if (RELU) v = fmaxf(v, 0.f); crow[nbase + 32 + lrow] = (OutT)v;
      v = a13[i] + bv3; if (RELU) v = fmaxf(v, 0.f); crow[nbase + 48 + lrow] = (OutT)v;
    }
  }
}

extern "C" void kernel_launch(void* const* d_in, const int* in_sizes, int n_in,
                              void* d_out, int out_size, void* d_ws, size_t ws_size,
                              hipStream_t stream) {
  const float* x_src      = (const float*)d_in[0];
  const float* x_dst      = (const float*)d_in[1];
  const int*   edge_index = (const int*)d_in[2];
  const int*   edge_type  = (const int*)d_in[3];
  const float* w1         = (const float*)d_in[4];
  const float* root1      = (const float*)d_in[5];
  const float* b1         = (const float*)d_in[6];
  const float* w2         = (const float*)d_in[7];
  const float* root2      = (const float*)d_in[8];
  const float* b2         = (const float*)d_in[9];
  float* out = (float*)d_out;

  const int E = in_sizes[3];        // num edges
  const int M = in_sizes[1] / D_CH; // num dst nodes

  // workspace layout (all offsets keep >=32B alignment)
  float* sums = (float*)d_ws;                          // M x 2048 f32
  float* cnt  = sums + (size_t)M * K0LEN;              // M x 8 f32
  bf16*  aggb = (bf16*)(cnt + (size_t)M * NREL);       // M x 2048 bf16
  bf16*  xdb  = aggb + (size_t)M * K0LEN;              // M x 256 bf16
  bf16*  hb   = xdb + (size_t)M * D_CH;                // M x 256 bf16
  bf16*  Bp1  = hb + (size_t)M * D_CH;                 // KTOT x 256 bf16 packed
  bf16*  Bp2  = Bp1 + (size_t)KTOT * D_CH;             // KTOT x 256 bf16 packed

  const size_t nz = (size_t)M * K0LEN + (size_t)M * NREL; // divisible by 4
  unsigned zb = (unsigned)((nz / 4 + 255) / 256);
  k_zero<<<zb, 256, 0, stream>>>(sums, (unsigned)nz);

  k_scatter<<<(E + 3) / 4, 256, 0, stream>>>(x_src, edge_index, edge_type, sums, cnt, E);

  const unsigned nseg = (unsigned)M * NREL;
  unsigned db = (unsigned)(((size_t)nseg * D_CH / 4 + 255) / 256);
  k_div_cvt<<<db, 256, 0, stream>>>(sums, cnt, aggb, nseg);

  const unsigned nxd = (unsigned)M * D_CH;
  k_cvt<<<(nxd / 4 + 255) / 256, 256, 0, stream>>>(x_dst, xdb, nxd);

  const unsigned pb = (KTOT * D_CH + 255) / 256; // 2304 blocks
  k_packB<<<pb, 256, 0, stream>>>(w1, root1, Bp1);
  k_packB<<<pb, 256, 0, stream>>>(w2, root2, Bp2);

  const int mblocks = (M + 31) / 32;
  k_gemm<1, bf16><<<mblocks, 128, 0, stream>>>(aggb, xdb, Bp1, b1, hb, M);
  k_gemm<0, float><<<mblocks, 128, 0, stream>>>(aggb, hb, Bp2, b2, out, M);
}
